// ContrastiveLoss_45011257262229
// MI455X (gfx1250) — matrix operand order repaired and checked by reading, not compiled, not run on hardware
//
#include <hip/hip_runtime.h>
#include <hip/hip_bf16.h>

typedef __attribute__((ext_vector_type(16))) __bf16 v16bf;
typedef __attribute__((ext_vector_type(8)))  __bf16 v8bf;
typedef __attribute__((ext_vector_type(8)))  float  v8f;
typedef int v4i_vs __attribute__((vector_size(16)));            // matches builtin param pointee
typedef __attribute__((address_space(1))) v4i_vs* gas_v4i;      // global (AS1) int4*
typedef __attribute__((address_space(3))) v4i_vs* las_v4i;      // LDS (AS3) int4*

#define N_ROWS 16384
#define D_DIM  256
#define NT     (N_ROWS / 16)          // 1024 column tiles
#define INV_T  (1.0f/0.07f)
#define LDS_STRIDE 264                // 256 + 8 pad: 4-bank advance/column -> conflict-free b128 reads

#if defined(__gfx1250__) && __has_builtin(__builtin_amdgcn_global_load_async_to_lds_b128) && __has_builtin(__builtin_amdgcn_s_wait_asynccnt)
#define HAVE_ASYNC 1
#else
#define HAVE_ASYNC 0
#endif

// ---------------- kernel 1: L2-normalize rows, split fp32 -> bf16 hi + bf16 lo ----------------
__global__ void nrm_split_kernel(const float* __restrict__ fl, const float* __restrict__ fg,
                                 __bf16* __restrict__ flh, __bf16* __restrict__ fll,
                                 __bf16* __restrict__ fgh, __bf16* __restrict__ fgl)
{
    const int row = blockIdx.x;          // 0 .. 2N-1 (first N = local, second N = global)
    const int t   = threadIdx.x;         // 256 threads = D
    const float* src;
    __bf16 *dh, *dl;
    if (row < N_ROWS) {
        src = fl + (size_t)row * D_DIM;
        dh  = flh + (size_t)row * D_DIM;
        dl  = fll + (size_t)row * D_DIM;
    } else {
        const int r = row - N_ROWS;
        src = fg + (size_t)r * D_DIM;
        dh  = fgh + (size_t)r * D_DIM;
        dl  = fgl + (size_t)r * D_DIM;
    }
    const float x = src[t];
    float v = x * x;
    #pragma unroll
    for (int m = 16; m >= 1; m >>= 1) v += __shfl_xor(v, m);
    __shared__ float part[8];
    if ((t & 31) == 0) part[t >> 5] = v;
    __syncthreads();
    float ss = 0.f;
    #pragma unroll
    for (int i = 0; i < 8; i++) ss += part[i];
    const float inv = 1.0f / fmaxf(sqrtf(ss), 1e-12f);
    const float y   = x * inv;
    const __bf16 h  = (__bf16)y;                 // RNE
    const __bf16 lo = (__bf16)(y - (float)h);    // residual
    dh[t] = h;
    dl[t] = lo;
}

// ---- stage one 16-row FG tile (hi+lo) into an LDS buffer ----
// thread t copies 16 elems (32B) of FG row (t>>4), chunk (t&15), for each of hi/lo.
__device__ __forceinline__ void stage_tile(const __bf16* __restrict__ fgh,
                                           const __bf16* __restrict__ fgl,
                                           __bf16* dh, __bf16* dl,
                                           int ct, int stRow, int stCol)
{
    const size_t g = (size_t)(ct * 16 + stRow) * D_DIM + stCol;
    const int    l = stRow * LDS_STRIDE + stCol;
#if HAVE_ASYNC
    __builtin_amdgcn_global_load_async_to_lds_b128((gas_v4i)(fgh + g),     (las_v4i)(dh + l),     0, 0);
    __builtin_amdgcn_global_load_async_to_lds_b128((gas_v4i)(fgh + g + 8), (las_v4i)(dh + l + 8), 0, 0);
    __builtin_amdgcn_global_load_async_to_lds_b128((gas_v4i)(fgl + g),     (las_v4i)(dl + l),     0, 0);
    __builtin_amdgcn_global_load_async_to_lds_b128((gas_v4i)(fgl + g + 8), (las_v4i)(dl + l + 8), 0, 0);
#else
    union F { v16bf v; v8bf h[2]; };
    F u;
    u.v = *(const v16bf*)(fgh + g);
    *(v8bf*)(dh + l)     = u.h[0];
    *(v8bf*)(dh + l + 8) = u.h[1];
    u.v = *(const v16bf*)(fgl + g);
    *(v8bf*)(dl + l)     = u.h[0];
    *(v8bf*)(dl + l + 8) = u.h[1];
#endif
}

// ---------------- kernel 2: WMMA GEMM + streaming logsumexp ----------------
// Each wave owns 16 local rows (A kept in VGPRs for the whole sweep).
// WG (8 waves = 128 rows) double-buffers 16-col FG tiles in LDS via async global->LDS copies.
__global__ __launch_bounds__(256) void ctr_main_kernel(
        const __bf16* __restrict__ flh, const __bf16* __restrict__ fll,
        const __bf16* __restrict__ fgh, const __bf16* __restrict__ fgl,
        float* __restrict__ rowRes)
{
    __shared__ __bf16 lb[2][2][16 * LDS_STRIDE];   // [buffer][hi/lo][tile]

    const int tid  = threadIdx.x;
    const int lane = tid & 31;
    const int wave = tid >> 5;
    const int half = lane >> 4;          // K-half selector for A / B fragments
    const int l15  = lane & 15;
    const int rowBase = blockIdx.x * 128 + wave * 16;

    // ---- preload A fragments (16 rows x 256 feats, hi+lo) into registers ----
    // ISA layout (16-bit A 16x32): lanes 0-15: K {8h+0..7} and {16+8h+0..7}, h = lane>>4
    union F { v16bf v; v8bf h[2]; };
    v16bf Ah[8], Al[8];
    {
        const size_t rb = (size_t)(rowBase + l15) * D_DIM;
        #pragma unroll
        for (int ks = 0; ks < 8; ks++) {
            const int kb = ks * 32 + half * 8;
            F u;
            u.h[0] = *(const v8bf*)(flh + rb + kb);
            u.h[1] = *(const v8bf*)(flh + rb + kb + 16);
            Ah[ks] = u.v;
            u.h[0] = *(const v8bf*)(fll + rb + kb);
            u.h[1] = *(const v8bf*)(fll + rb + kb + 16);
            Al[ks] = u.v;
        }
    }

    float S[8], pos[8];
    #pragma unroll
    for (int e = 0; e < 8; e++) { S[e] = 0.f; pos[e] = 0.f; }

    const int stRow = tid >> 4;          // staging role
    const int stCol = (tid & 15) * 16;

    // prologue: stage tile 0 into buffer 0
    stage_tile(fgh, fgl, lb[0][0], lb[0][1], 0, stRow, stCol);

    #pragma unroll 2
    for (int ct = 0; ct < NT; ct++) {
        // issue next tile's async copies into the other buffer (wraps on last iter,
        // keeping the wait immediate constant; the overwritten buffer was fenced
        // by the previous iteration's trailing barrier)
        const int nxt = (ct + 1) & (NT - 1);
        stage_tile(fgh, fgl, lb[(ct + 1) & 1][0], lb[(ct + 1) & 1][1], nxt, stRow, stCol);

#if HAVE_ASYNC
        __builtin_amdgcn_s_wait_asynccnt(4);   // async ops complete in order: tile ct landed
#endif
        __syncthreads();                       // all waves' portions visible

        const __bf16* bhp = lb[ct & 1][0];
        const __bf16* blp = lb[ct & 1][1];

        // ---- 16x16 sim tile: bf16x3 compensated GEMM, fp32 accumulate ----
        v8f acc = {};
        #pragma unroll
        for (int ks = 0; ks < 8; ks++) {
            // B layout (32x16 16-bit): lane n=l15 holds K = 32*ks + {8*half + 0..7, 16 + 8*half + 0..7}
            const int bo = l15 * LDS_STRIDE + ks * 32 + half * 8;
            F bh, bl;
            bh.h[0] = *(const v8bf*)(bhp + bo);
            bh.h[1] = *(const v8bf*)(bhp + bo + 16);
            bl.h[0] = *(const v8bf*)(blp + bo);
            bl.h[1] = *(const v8bf*)(blp + bo + 16);
            acc = __builtin_amdgcn_wmma_f32_16x16x32_bf16(false, Ah[ks], false, bh.v, (short)0, acc, false, false);
            acc = __builtin_amdgcn_wmma_f32_16x16x32_bf16(false, Ah[ks], false, bl.v, (short)0, acc, false, false);
            acc = __builtin_amdgcn_wmma_f32_16x16x32_bf16(false, Al[ks], false, bh.v, (short)0, acc, false, false);
        }

        // ---- streaming sum(exp) — sims bounded by 1/T=14.3 so no max-shift needed ----
        const int col = ct * 16 + l15;
        #pragma unroll
        for (int e = 0; e < 8; e++) {
            const float x = acc[e] * INV_T;    // C/D layout: row = e + 8*half, col = l15
            S[e] += __expf(x);
            const int row = rowBase + 8 * half + e;
            if (row == col) pos[e] = x;
        }

        __syncthreads();   // all waves done reading buf[ct&1] before it is re-staged
    }

    // ---- reduce across the 16 lanes holding each row; write per-row loss ----
    #pragma unroll
    for (int e = 0; e < 8; e++) {
        float s = S[e], p = pos[e];
        #pragma unroll
        for (int m = 8; m >= 1; m >>= 1) { s += __shfl_xor(s, m); p += __shfl_xor(p, m); }
        if (l15 == 0) rowRes[rowBase + 8 * half + e] = __logf(s) - p;
    }
}

// ---------------- kernel 3: mean over rows ----------------
__global__ void reduce_mean_kernel(const float* __restrict__ rowRes, float* __restrict__ out)
{
    __shared__ float part[8];
    const int t = threadIdx.x;
    float s = 0.f;
    for (int i = t; i < N_ROWS; i += 256) s += rowRes[i];
    #pragma unroll
    for (int m = 16; m >= 1; m >>= 1) s += __shfl_xor(s, m);
    if ((t & 31) == 0) part[t >> 5] = s;
    __syncthreads();
    if (t == 0) {
        float tot = 0.f;
        #pragma unroll
        for (int i = 0; i < 8; i++) tot += part[i];
        out[0] = tot / (float)N_ROWS;
    }
}

extern "C" void kernel_launch(void* const* d_in, const int* in_sizes, int n_in,
                              void* d_out, int out_size, void* d_ws, size_t ws_size,
                              hipStream_t stream)
{
    const float* fl = (const float*)d_in[0];
    const float* fg = (const float*)d_in[1];
    char* ws = (char*)d_ws;
    const size_t NE = (size_t)N_ROWS * D_DIM;   // 4,194,304 elems

    __bf16* flh = (__bf16*)(ws);
    __bf16* fll = (__bf16*)(ws + NE * 2);
    __bf16* fgh = (__bf16*)(ws + NE * 4);
    __bf16* fgl = (__bf16*)(ws + NE * 6);
    float*  rowRes = (float*)(ws + NE * 8);     // 64 KB after the 32 MB bf16 region

    hipLaunchKernelGGL(nrm_split_kernel, dim3(2 * N_ROWS), dim3(256), 0, stream,
                       fl, fg, flh, fll, fgh, fgl);
    hipLaunchKernelGGL(ctr_main_kernel, dim3(N_ROWS / 128), dim3(256), 0, stream,
                       flh, fll, fgh, fgl, rowRes);
    hipLaunchKernelGGL(reduce_mean_kernel, dim3(1), dim3(256), 0, stream,
                       rowRes, (float*)d_out);
}
